// AHGNN_61735859913301
// MI455X (gfx1250) — matrix operand order repaired and checked by reading, not compiled, not run on hardware
//
#include <hip/hip_runtime.h>
#include <hip/hip_bf16.h>
#include <math.h>

#define BQ 4
#define NQ 16384
#define MQ 512
#define CQ 256
#define KQ 24
#define BN_TOTAL (BQ * NQ)
#define AROW 260                      // LDS row stride (floats): 256 + 4 pad -> bank-conflict-free

typedef __attribute__((ext_vector_type(16))) __bf16        v16bf;
typedef __attribute__((ext_vector_type(8)))  float         v8f;
typedef __attribute__((ext_vector_type(4)))  unsigned int  v4u;
typedef __attribute__((ext_vector_type(8)))  int           v8i;
typedef __attribute__((ext_vector_type(4)))  int           v4i;

#if __has_builtin(__builtin_amdgcn_tensor_load_to_lds)
  #define HAVE_TDM 1
  #if __has_include(<hip/amd_detail/amd_gfx1250_TDM.h>)
    #define TDM_6ARG 1
  #endif
#endif

__device__ __forceinline__ unsigned ahg_lds_offset(const void* p) {
    // generic -> LDS(addrspace 3) explicit cast yields the 32-bit LDS byte offset
    return (unsigned)(unsigned long long)(const __attribute__((address_space(3))) void*)p;
}

// ---------------------------------------------------------------------------
// 0) zero scratch accumulators (E, deg_e, s1, s2 are contiguous)
// ---------------------------------------------------------------------------
__global__ void ahg_zero_kernel(float* __restrict__ p, int n) {
    int i = blockIdx.x * 256 + threadIdx.x;
    if (i < n) p[i] = 0.0f;
}

// ---------------------------------------------------------------------------
// 1) one-time weight prep: fc_w -> WMMA-fragment-ordered bf16 hi/lo buffers
//    fragment (ct, kbi, lane) holds 16 bf16: B[k][n], n = lane%16,
//    k = kbi*32 + (lane<16 ? 0 : 16) + j   (contiguous k order, same as A fragment)
//    4096 threads total.
// ---------------------------------------------------------------------------
__global__ void ahg_wprep_kernel(const float* __restrict__ w,
                                 __bf16* __restrict__ whi,
                                 __bf16* __restrict__ wlo) {
    const int t    = blockIdx.x * 256 + threadIdx.x;   // 0..4095
    const int lane = t & 31;
    const int kbi  = (t >> 5) & 7;
    const int ct   = t >> 8;                           // col tile 0..15
    const int l16  = lane & 15;
    const int half = lane >> 4;
    const int col  = ct * 16 + l16;
    const int kb   = kbi * 32 + (half ? 16 : 0);
    const float* __restrict__ src = w + (size_t)col * CQ + kb;
    const size_t base = (size_t)t * 16;
#pragma unroll
    for (int j = 0; j < 16; ++j) {
        const float v  = src[j];
        const __bf16 h = (__bf16)v;
        whi[base + j] = h;
        wlo[base + j] = (__bf16)(v - (float)h);
    }
}

// ---------------------------------------------------------------------------
// 2) top-K nearest anchors per point; also count hyperedge degrees
// ---------------------------------------------------------------------------
__global__ void ahg_topk_kernel(const float* __restrict__ coords,
                                const float* __restrict__ anchors,
                                int* __restrict__ idx,
                                float* __restrict__ deg_e) {
    __shared__ float sa[MQ * 3];
    const int b    = blockIdx.x / (NQ / 256);
    const int nblk = blockIdx.x % (NQ / 256);
    for (int i = threadIdx.x; i < MQ * 3; i += 256)
        sa[i] = anchors[b * MQ * 3 + i];
    __syncthreads();

    const int n = nblk * 256 + threadIdx.x;
    const float cx = coords[(b * NQ + n) * 3 + 0];
    const float cy = coords[(b * NQ + n) * 3 + 1];
    const float cz = coords[(b * NQ + n) * 3 + 2];

    float bd[KQ];
    int   bi[KQ];
#pragma unroll
    for (int k = 0; k < KQ; ++k) { bd[k] = 3.4e38f; bi[k] = 0; }

    for (int m = 0; m < MQ; ++m) {
        const float dx = cx - sa[m * 3 + 0];
        const float dy = cy - sa[m * 3 + 1];
        const float dz = cz - sa[m * 3 + 2];
        const float d  = dx * dx + dy * dy + dz * dz;
        if (d < bd[KQ - 1]) {
            int p = KQ - 1;
            while (p > 0 && bd[p - 1] > d) {
                bd[p] = bd[p - 1]; bi[p] = bi[p - 1]; --p;
            }
            bd[p] = d; bi[p] = m;
        }
    }
#pragma unroll
    for (int k = 0; k < KQ; ++k) {
        idx[((size_t)(b * NQ + n)) * KQ + k] = bi[k];
        unsafeAtomicAdd(&deg_e[b * MQ + bi[k]], 1.0f);
    }
}

// ---------------------------------------------------------------------------
// 3) xf = x @ fc_w^T + fc_b, bf16x3 split-precision WMMA (f32 accumulate).
//    One block per 16-row strip; A tile (16x256, contiguous 16KB of x) staged
//    to LDS by the Tensor Data Mover with row padding (stride 260 floats).
//    8 waves x 2 col-tiles cover all 256 output columns.
// ---------------------------------------------------------------------------
__global__ void ahg_gemm_kernel(const float* __restrict__ x,
                                const __bf16* __restrict__ whi,
                                const __bf16* __restrict__ wlo,
                                const float* __restrict__ bias,
                                float* __restrict__ xf) {
    __shared__ float lds_x[16 * AROW];

    const int wave = threadIdx.x >> 5;
    const int lane = threadIdx.x & 31;
    const int half = lane >> 4;
    const int l16  = lane & 15;
    const int row0 = blockIdx.x * 16;

#if defined(HAVE_TDM)
    if (wave == 0) {
        const unsigned lds_base = ahg_lds_offset(lds_x);
        const unsigned long long ga =
            (unsigned long long)(x + (size_t)row0 * CQ);
        // D# group0: count=1 | lds_addr | global_addr | type=2
        const v4u g0 = { 1u, lds_base, (unsigned)ga,
                         (unsigned)((ga >> 32) & 0x1FFFFFFu) | 0x80000000u };
        // D# group1: data_size=4B(code2), pad_enable, pad every 256 dw (code7),
        //            pad 4 dw (code3); 1-D tile of 4096 dwords, stride 4096.
        const v8i g1 = { (int)0x07D20000,
                         (int)(4096u << 16),   // tensor_dim0 low16 (<<16)
                         (int)(1u << 16),      // tensor_dim1 low16 (<<16)
                         (int)(4096u << 16),   // tile_dim0 (<<16)
                         0,                    // tile_dim1=0 (1-D), tile_dim2=0
                         4096,                 // tensor_dim0_stride low32
                         0, 0 };
        const v4i g2 = { 0, 0, 0, 0 };
        const v4i g3 = { 0, 0, 0, 0 };
#if defined(TDM_6ARG)
        const v8i g4 = { 0, 0, 0, 0, 0, 0, 0, 0 };
        __builtin_amdgcn_tensor_load_to_lds(g0, g1, g2, g3, g4, 0);
#else
        __builtin_amdgcn_tensor_load_to_lds(g0, g1, g2, g3, 0);
#endif
        __builtin_amdgcn_s_wait_tensorcnt(0);
    }
#else
    for (int i = threadIdx.x; i < 16 * CQ; i += 256) {
        const int r = i >> 8, c = i & 255;
        lds_x[r * AROW + c] = x[(size_t)row0 * CQ + i];
    }
#endif
    __syncthreads();

    const int ct0 = wave;        // column tiles this wave owns
    const int ct1 = wave + 8;
    v8f acc0 = {}, acc1 = {};

    for (int kbi = 0; kbi < 8; ++kbi) {
        const int kb = kbi * 32;
        // --- A fragment from LDS (contiguous K: k = 16*half + j, matches B prep), hi/lo ---
        v16bf a_hi, a_lo;
#pragma unroll
        for (int j = 0; j < 16; ++j) {
            const int kk = (half ? 16 : 0) + j;
            const float v  = lds_x[l16 * AROW + kb + kk];
            const __bf16 h = (__bf16)v;
            a_hi[j] = h;
            a_lo[j] = (__bf16)(v - (float)h);
        }
        // --- pre-converted B fragments: coalesced 32B vector loads ---
        const v16bf bh0 = *(const v16bf*)(whi + ((size_t)(ct0 * 8 + kbi) * 32 + lane) * 16);
        const v16bf bl0 = *(const v16bf*)(wlo + ((size_t)(ct0 * 8 + kbi) * 32 + lane) * 16);
        const v16bf bh1 = *(const v16bf*)(whi + ((size_t)(ct1 * 8 + kbi) * 32 + lane) * 16);
        const v16bf bl1 = *(const v16bf*)(wlo + ((size_t)(ct1 * 8 + kbi) * 32 + lane) * 16);

        acc0 = __builtin_amdgcn_wmma_f32_16x16x32_bf16(false, a_lo, false, bh0,
                                                       (short)0, acc0, false, false);
        acc0 = __builtin_amdgcn_wmma_f32_16x16x32_bf16(false, a_hi, false, bl0,
                                                       (short)0, acc0, false, false);
        acc0 = __builtin_amdgcn_wmma_f32_16x16x32_bf16(false, a_hi, false, bh0,
                                                       (short)0, acc0, false, false);
        acc1 = __builtin_amdgcn_wmma_f32_16x16x32_bf16(false, a_lo, false, bh1,
                                                       (short)0, acc1, false, false);
        acc1 = __builtin_amdgcn_wmma_f32_16x16x32_bf16(false, a_hi, false, bl1,
                                                       (short)0, acc1, false, false);
        acc1 = __builtin_amdgcn_wmma_f32_16x16x32_bf16(false, a_hi, false, bh1,
                                                       (short)0, acc1, false, false);
    }

    const float bv0 = bias[ct0 * 16 + l16];
    const float bv1 = bias[ct1 * 16 + l16];
#pragma unroll
    for (int r = 0; r < 8; ++r) {
        const int mrow = r + (half ? 8 : 0);      // C/D layout: VGPR r -> row
        xf[(size_t)(row0 + mrow) * CQ + ct0 * 16 + l16] = acc0[r] + bv0;
        xf[(size_t)(row0 + mrow) * CQ + ct1 * 16 + l16] = acc1[r] + bv1;
    }
}

// ---------------------------------------------------------------------------
// 4) scatter xf rows into hyperedge accumulators E (L2-resident f32 atomics)
// ---------------------------------------------------------------------------
__global__ void ahg_scatter_kernel(const float* __restrict__ xf,
                                   const int* __restrict__ idx,
                                   float* __restrict__ E) {
    const int lane = threadIdx.x & 31;
    const int wave = threadIdx.x >> 5;
    const int p    = blockIdx.x * 8 + wave;   // 0..BN_TOTAL-1
    const int b    = p >> 14;                 // / NQ

    float v[8];
    const float* __restrict__ row = xf + (size_t)p * CQ;
#pragma unroll
    for (int j = 0; j < 8; ++j) v[j] = row[lane + 32 * j];

    const int* __restrict__ ip = idx + (size_t)p * KQ;
    float* __restrict__ Eb = E + (size_t)b * MQ * CQ;
#pragma unroll 4
    for (int k = 0; k < KQ; ++k) {
        float* __restrict__ em = Eb + (size_t)ip[k] * CQ;
#pragma unroll
        for (int j = 0; j < 8; ++j)
            unsafeAtomicAdd(&em[lane + 32 * j], v[j]);
    }
}

// ---------------------------------------------------------------------------
// 5) normalize hyperedges by degree (mean aggregation)
// ---------------------------------------------------------------------------
__global__ void ahg_norme_kernel(float* __restrict__ E,
                                 const float* __restrict__ deg) {
    const int i  = blockIdx.x * 256 + threadIdx.x;  // BQ*MQ*CQ total
    const int bm = i / CQ;
    const float d   = deg[bm];
    const float inv = d > 0.0f ? 1.0f / d : 0.0f;
    E[i] *= inv;
}

// ---------------------------------------------------------------------------
// 6) gather e->v mean + residual; fused per-channel sum / sumsq for BN
// ---------------------------------------------------------------------------
__global__ void ahg_gather_kernel(const float* __restrict__ x,
                                  const float* __restrict__ E,
                                  const int* __restrict__ idx,
                                  float* __restrict__ pre,
                                  float* __restrict__ s1,
                                  float* __restrict__ s2) {
    __shared__ float ls1[CQ];
    __shared__ float ls2[CQ];
    for (int i = threadIdx.x; i < CQ; i += 256) { ls1[i] = 0.0f; ls2[i] = 0.0f; }
    __syncthreads();

    const int lane = threadIdx.x & 31;
    const int wave = threadIdx.x >> 5;
    float a1[8] = {}, a2[8] = {};

    for (int t = 0; t < 16; ++t) {
        const int p = blockIdx.x * 128 + wave * 16 + t;
        const int b = p >> 14;
        const int* __restrict__ ip = idx + (size_t)p * KQ;
        const float* __restrict__ Eb = E + (size_t)b * MQ * CQ;
        float acc[8] = {};
#pragma unroll 4
        for (int k = 0; k < KQ; ++k) {
            const float* __restrict__ em = Eb + (size_t)ip[k] * CQ;
#pragma unroll
            for (int j = 0; j < 8; ++j) acc[j] += em[lane + 32 * j];
        }
        const float* __restrict__ xr = x + (size_t)p * CQ;
        float* __restrict__ pr = pre + (size_t)p * CQ;
#pragma unroll
        for (int j = 0; j < 8; ++j) {
            const float val = xr[lane + 32 * j] + acc[j] * (1.0f / (float)KQ);
            pr[lane + 32 * j] = val;
            a1[j] += val;
            a2[j] += val * val;
        }
    }
#pragma unroll
    for (int j = 0; j < 8; ++j) {
        atomicAdd(&ls1[lane + 32 * j], a1[j]);   // ds_add_f32
        atomicAdd(&ls2[lane + 32 * j], a2[j]);
    }
    __syncthreads();
    for (int i = threadIdx.x; i < CQ; i += 256) {
        unsafeAtomicAdd(&s1[i], ls1[i]);
        unsafeAtomicAdd(&s2[i], ls2[i]);
    }
}

// ---------------------------------------------------------------------------
// 7) LDS-tiled transpose [B,N,C] -> [B,C,N] fused with BatchNorm + SiLU
// ---------------------------------------------------------------------------
__global__ void ahg_final_kernel(const float* __restrict__ pre,
                                 const float* __restrict__ s1,
                                 const float* __restrict__ s2,
                                 const float* __restrict__ gamma,
                                 const float* __restrict__ beta,
                                 float* __restrict__ out) {
    __shared__ float tile[32][33];
    const int b  = blockIdx.z;
    const int n0 = blockIdx.x * 32;
    const int c0 = blockIdx.y * 32;
    const int tx = threadIdx.x;
    const int ty = threadIdx.y;

#pragma unroll
    for (int j = 0; j < 4; ++j) {
        const int n = n0 + ty + 8 * j;
        tile[ty + 8 * j][tx] = pre[((size_t)(b * NQ + n)) * CQ + c0 + tx];
    }
    __syncthreads();

    const float invBN = 1.0f / (float)BN_TOTAL;
#pragma unroll
    for (int j = 0; j < 4; ++j) {
        const int c = c0 + ty + 8 * j;
        const float mean = s1[c] * invBN;
        const float var  = s2[c] * invBN - mean * mean;
        const float rstd = rsqrtf(var + 1e-5f);
        const float g = gamma[c], be = beta[c];
        const float v = tile[tx][ty + 8 * j];
        const float y = (v - mean) * rstd * g + be;
        const float silu = y / (1.0f + __expf(-y));
        out[((size_t)(b * CQ + c)) * NQ + (n0 + tx)] = silu;
    }
}

// ---------------------------------------------------------------------------
// launch
// ---------------------------------------------------------------------------
extern "C" void kernel_launch(void* const* d_in, const int* in_sizes, int n_in,
                              void* d_out, int out_size, void* d_ws, size_t ws_size,
                              hipStream_t stream) {
    (void)in_sizes; (void)n_in; (void)out_size; (void)ws_size;

    const float* x       = (const float*)d_in[0];
    const float* coords  = (const float*)d_in[1];
    const float* anchors = (const float*)d_in[2];
    const float* fc_w    = (const float*)d_in[3];
    const float* fc_b    = (const float*)d_in[4];
    const float* gamma   = (const float*)d_in[5];
    const float* beta    = (const float*)d_in[6];
    float* out = (float*)d_out;

    // workspace (floats): xf/pre | E | deg_e | s1 | s2 | idx(int) | whi | wlo
    float* ws  = (float*)d_ws;
    float* xf  = ws;                                   // BN_TOTAL*CQ (reused as pre)
    float* E   = ws + (size_t)BN_TOTAL * CQ;           // BQ*MQ*CQ
    float* deg = E + (size_t)BQ * MQ * CQ;             // BQ*MQ
    float* s1  = deg + BQ * MQ;                        // CQ
    float* s2  = s1 + CQ;                              // CQ
    int*   idx = (int*)(s2 + CQ);                      // BN_TOTAL*KQ
    __bf16* whi = (__bf16*)(idx + (size_t)BN_TOTAL * KQ); // 65536 bf16
    __bf16* wlo = whi + 16 * 8 * 32 * 16;                 // 65536 bf16
    float* pre = xf;                                   // alias: xf dead after scatter

    // 0) zero accumulators (E, deg, s1, s2 contiguous)
    const int nzero = BQ * MQ * CQ + BQ * MQ + 2 * CQ;
    ahg_zero_kernel<<<(nzero + 255) / 256, 256, 0, stream>>>(E, nzero);

    // 1) one-time weight fragment prep (bf16 hi/lo)
    ahg_wprep_kernel<<<16, 256, 0, stream>>>(fc_w, whi, wlo);

    // 2) top-K + hyperedge degrees
    ahg_topk_kernel<<<BQ * (NQ / 256), 256, 0, stream>>>(coords, anchors, idx, deg);

    // 3) WMMA GEMM with TDM-staged A tiles
    ahg_gemm_kernel<<<BN_TOTAL / 16, 256, 0, stream>>>(x, whi, wlo, fc_b, xf);

    // 4) scatter v->e
    ahg_scatter_kernel<<<BN_TOTAL / 8, 256, 0, stream>>>(xf, idx, E);

    // 5) hyperedge mean
    ahg_norme_kernel<<<(BQ * MQ * CQ) / 256, 256, 0, stream>>>(E, deg);

    // 6) gather e->v mean + residual + BN stats (writes pre over xf)
    ahg_gather_kernel<<<BN_TOTAL / 128, 256, 0, stream>>>(x, E, idx, pre, s1, s2);

    // 7) transpose + BN + SiLU
    ahg_final_kernel<<<dim3(NQ / 32, CQ / 32, BQ), dim3(32, 8), 0, stream>>>(
        pre, s1, s2, gamma, beta, out);
}
